// Aligner_70368744178365
// MI455X (gfx1250) — matrix operand order repaired and checked by reading, not compiled, not run on hardware
//
#include <hip/hip_runtime.h>
#include <hip/hip_bf16.h>
#include <math.h>
#include <stdint.h>

#define B_ 16
#define S_ 1024
#define D_ 1024

typedef __attribute__((ext_vector_type(16))) __bf16 v16bf;
typedef __attribute__((ext_vector_type(8)))  __bf16 v8bf;
typedef __attribute__((ext_vector_type(8)))  float  v8f;
typedef __attribute__((ext_vector_type(4)))  float  v4f;

// ---------------------------------------------------------------------------
// CDNA5 async copy: global -> LDS, 16 bytes per lane, tracked by ASYNCcnt.
// lds_addr = low 32 bits of the flat LDS address (aperture keeps offset there).
// ---------------------------------------------------------------------------
__device__ __forceinline__ void async_copy_b128(uint32_t lds_addr, const void* gptr) {
  asm volatile("global_load_async_to_lds_b128 %0, %1, off"
               :: "v"(lds_addr), "v"((unsigned long long)(uintptr_t)gptr)
               : "memory");
}
#define WAIT_ASYNCCNT_LE4() asm volatile("s_wait_asynccnt 0x4" ::: "memory")
#define WAIT_ASYNCCNT_0()   asm volatile("s_wait_asynccnt 0x0" ::: "memory")

__device__ __forceinline__ uint32_t lds_addr32(const void* p) {
  return (uint32_t)(uintptr_t)p;
}

// ---------------------------------------------------------------------------
// Kernel 0: f32 -> bf16 conversion (one shot, so the GEMM can async-copy raw
// bf16 tiles straight into LDS with no VALU in the inner loop)
// ---------------------------------------------------------------------------
__global__ __launch_bounds__(256) void cvt_f32_bf16_kernel(const float* __restrict__ in,
                                                           __bf16* __restrict__ out) {
  const size_t i = ((size_t)blockIdx.x * 256 + threadIdx.x) * 8;
  const v4f a = *(const v4f*)(in + i);
  const v4f b = *(const v4f*)(in + i + 4);
  v8bf o;
  o[0] = (__bf16)a.x; o[1] = (__bf16)a.y; o[2] = (__bf16)a.z; o[3] = (__bf16)a.w;
  o[4] = (__bf16)b.x; o[5] = (__bf16)b.y; o[6] = (__bf16)b.z; o[7] = (__bf16)b.w;
  *(v8bf*)(out + i) = o;
}

// ---------------------------------------------------------------------------
// Kernel 1: special-token mask -> additive mask (-10000) and usable lengths
// ---------------------------------------------------------------------------
__global__ __launch_bounds__(256) void mask_len_kernel(const int* __restrict__ ids,
                                                       float* __restrict__ addm,
                                                       float* __restrict__ lenv) {
  const int b = blockIdx.x;
  __shared__ float red[256];
  float c = 0.0f;
  for (int i = threadIdx.x; i < S_; i += 256) {
    const int id = ids[b * S_ + i];
    const bool sp = (id == 0) || (id == 101) || (id == 102);  // PAD/CLS/SEP
    addm[b * S_ + i] = sp ? -10000.0f : 0.0f;
    c += sp ? 0.0f : 1.0f;
  }
  red[threadIdx.x] = c;
  __syncthreads();
  for (int st = 128; st > 0; st >>= 1) {
    if (threadIdx.x < st) red[threadIdx.x] += red[threadIdx.x + st];
    __syncthreads();
  }
  if (threadIdx.x == 0) lenv[b] = red[0];
}

// ---------------------------------------------------------------------------
// Kernel 2: batched scores = h_src @ h_tgt^T, v_wmma_f32_16x16x32_bf16.
// Block tile 128x128, 8 waves, wave = 2x4 fragments. Tiles are staged with
// GLOBAL_LOAD_ASYNC_TO_LDS_B128 and double-buffered on ASYNCcnt.
// ---------------------------------------------------------------------------
#define BM 128
#define BN 128
#define BK 32
#define LDK 40                       // padded LDS row stride (bf16 elems)
#define TILE_ELEMS (128 * LDK)       // one matrix tile in one buffer

__global__ __launch_bounds__(256) void gemm_scores_wmma(const __bf16* __restrict__ A,
                                                        const __bf16* __restrict__ Bm,
                                                        float* __restrict__ C) {
  __shared__ __bf16 lds[2][2 * TILE_ELEMS];   // [buffer][A tile | B tile]

  const int b = blockIdx.z;
  const int mbase = blockIdx.y * BM;
  const int nbase = blockIdx.x * BN;
  const __bf16* Ab = A + (size_t)b * S_ * D_;
  const __bf16* Bb = Bm + (size_t)b * S_ * D_;
  float* Cb = C + (size_t)b * S_ * S_;

  const int tid = threadIdx.x;
  const int lane = tid & 31;
  const int wave = tid >> 5;
  const int wm = wave & 3;   // 4 row strips of 32
  const int wn = wave >> 2;  // 2 col strips of 64

  v8f acc[2][4] = {};

  // Staging map: 512 16-byte chunks per matrix tile (128 rows x 32 K bf16),
  // 2 chunks per thread per matrix. row = chunk>>2, q = chunk&3 (8 bf16 each).
  uint32_t ldsA_off[2], ldsB_off[2];      // byte offsets within one buffer
  const __bf16 *gA[2], *gB[2];
#pragma unroll
  for (int it = 0; it < 2; ++it) {
    const int c = tid + it * 256;
    const int row = c >> 2;
    const int q = c & 3;
    ldsA_off[it] = (uint32_t)((row * LDK + q * 8) * 2);
    ldsB_off[it] = (uint32_t)((TILE_ELEMS + row * LDK + q * 8) * 2);
    gA[it] = Ab + (size_t)(mbase + row) * D_ + q * 8;
    gB[it] = Bb + (size_t)(nbase + row) * D_ + q * 8;
  }
  const uint32_t ldsBase  = lds_addr32(&lds[0][0]);
  const uint32_t bufBytes = 2 * TILE_ELEMS * 2;

  auto stage = [&](int buf, int kb) {
    const uint32_t bo = ldsBase + (uint32_t)buf * bufBytes;
#pragma unroll
    for (int it = 0; it < 2; ++it) {
      async_copy_b128(bo + ldsA_off[it], gA[it] + kb);
      async_copy_b128(bo + ldsB_off[it], gB[it] + kb);
    }
  };

  const int laneM = lane & 15;
  const int aKoff = (lane < 16) ? 0 : 8;    // A: ISA K split {0..7,16..23}/{8..15,24..31}
  const int bKoff = (lane < 16) ? 0 : 8;    // B: ISA K split {0..7,16..23}/{8..15,24..31}

  stage(0, 0);                              // prime the pipeline (4 async ops)

  for (int kt = 0; kt < D_ / BK; ++kt) {
    const int cur = kt & 1;
    if (kt + 1 < D_ / BK) {
      stage(cur ^ 1, (kt + 1) * BK);        // 4 more in flight
      WAIT_ASYNCCNT_LE4();                  // current tile's 4 copies done
    } else {
      WAIT_ASYNCCNT_0();
    }
    __syncthreads();                        // everyone's copies visible

    const __bf16* tA = &lds[cur][0];
    const __bf16* tB = &lds[cur][TILE_ELEMS];

    v16bf afrag[2], bfrag[4];
#pragma unroll
    for (int i = 0; i < 2; ++i) {
      const int row = wm * 32 + i * 16 + laneM;
      const v8bf lo = *(const v8bf*)&tA[row * LDK + aKoff];
      const v8bf hi = *(const v8bf*)&tA[row * LDK + aKoff + 16];
#pragma unroll
      for (int e = 0; e < 8; ++e) { afrag[i][e] = lo[e]; afrag[i][8 + e] = hi[e]; }
    }
#pragma unroll
    for (int j = 0; j < 4; ++j) {
      const int col = wn * 64 + j * 16 + laneM;
      const v8bf lo = *(const v8bf*)&tB[col * LDK + bKoff];
      const v8bf hi = *(const v8bf*)&tB[col * LDK + bKoff + 16];
#pragma unroll
      for (int e = 0; e < 8; ++e) { bfrag[j][e] = lo[e]; bfrag[j][8 + e] = hi[e]; }
    }
#pragma unroll
    for (int i = 0; i < 2; ++i)
#pragma unroll
      for (int j = 0; j < 4; ++j)
        acc[i][j] = __builtin_amdgcn_wmma_f32_16x16x32_bf16(
            false, afrag[i], false, bfrag[j], (short)0, acc[i][j], false, false);

    __syncthreads();                        // done reading cur before overwrite
  }

  // C fragment layout: VGPR r -> M=r (lanes 0-15) / M=r+8 (lanes 16-31), N=lane&15
  const int rowOff = (lane < 16) ? 0 : 8;
  const int colL = lane & 15;
#pragma unroll
  for (int i = 0; i < 2; ++i)
#pragma unroll
    for (int j = 0; j < 4; ++j) {
      const int col = nbase + wn * 64 + j * 16 + colL;
#pragma unroll
      for (int r = 0; r < 8; ++r) {
        const int row = mbase + wm * 32 + i * 16 + r + rowOff;
        Cb[(size_t)row * S_ + col] = acc[i][j][r];
      }
    }
}

// ---------------------------------------------------------------------------
// Kernel 3: per-row (softmax over t) stats: max, sum(exp), sum(exp(/tau))
// ---------------------------------------------------------------------------
__global__ __launch_bounds__(256) void row_stats_kernel(const float* __restrict__ scores,
                                                        const float* __restrict__ add_tgt,
                                                        const float* __restrict__ len_tgt,
                                                        float* __restrict__ rstat) {
  const int b = blockIdx.x >> 10;
  const int s = blockIdx.x & (S_ - 1);
  const float* row = scores + ((size_t)b * S_ + s) * S_;
  const float* am = add_tgt + b * S_;
  __shared__ float red[256];

  float x[4];
  float m = -3.4e38f;
#pragma unroll
  for (int i = 0; i < 4; ++i) {
    const int t = threadIdx.x + i * 256;
    x[i] = row[t] + am[t];
    m = fmaxf(m, x[i]);
  }
  red[threadIdx.x] = m;
  __syncthreads();
  for (int st = 128; st > 0; st >>= 1) {
    if (threadIdx.x < st) red[threadIdx.x] = fmaxf(red[threadIdx.x], red[threadIdx.x + st]);
    __syncthreads();
  }
  m = red[0];
  __syncthreads();

  const float inv_tau = rsqrtf(len_tgt[b]);
  float s1 = 0.0f, sT = 0.0f;
#pragma unroll
  for (int i = 0; i < 4; ++i) {
    const float d = x[i] - m;
    s1 += __expf(d);
    sT += __expf(d * inv_tau);
  }
  red[threadIdx.x] = s1;
  __syncthreads();
  for (int st = 128; st > 0; st >>= 1) {
    if (threadIdx.x < st) red[threadIdx.x] += red[threadIdx.x + st];
    __syncthreads();
  }
  s1 = red[0];
  __syncthreads();
  red[threadIdx.x] = sT;
  __syncthreads();
  for (int st = 128; st > 0; st >>= 1) {
    if (threadIdx.x < st) red[threadIdx.x] += red[threadIdx.x + st];
    __syncthreads();
  }
  if (threadIdx.x == 0) {
    const int idx = (b * S_ + s) * 3;
    rstat[idx + 0] = m;
    rstat[idx + 1] = s1;
    rstat[idx + 2] = red[0];
  }
}

// ---------------------------------------------------------------------------
// Kernel 4: per-column (softmax over s) stats, coalesced 32-col scan groups
// ---------------------------------------------------------------------------
__global__ void col_stats_kernel(const float* __restrict__ scores,
                                 const float* __restrict__ add_src,
                                 const float* __restrict__ len_src,
                                 float* __restrict__ cstat) {
  const int b = blockIdx.y;
  const int col = blockIdx.x * 32 + threadIdx.x;
  const float* base = scores + (size_t)b * S_ * S_ + col;
  const float* am = add_src + b * S_;
  __shared__ float red[8][32];

  float m = -3.4e38f;
  for (int s = threadIdx.y; s < S_; s += 8) {
    __builtin_prefetch(base + (size_t)(s + 64) * S_, 0, 0);
    m = fmaxf(m, base[(size_t)s * S_] + am[s]);
  }
  red[threadIdx.y][threadIdx.x] = m;
  __syncthreads();
  if (threadIdx.y == 0) {
#pragma unroll
    for (int y = 1; y < 8; ++y) m = fmaxf(m, red[y][threadIdx.x]);
    red[0][threadIdx.x] = m;
  }
  __syncthreads();
  m = red[0][threadIdx.x];
  __syncthreads();

  const float inv_tau = rsqrtf(len_src[b]);
  float s1 = 0.0f, sT = 0.0f;
  for (int s = threadIdx.y; s < S_; s += 8) {
    const float d = base[(size_t)s * S_] + am[s] - m;
    s1 += __expf(d);
    sT += __expf(d * inv_tau);
  }
  red[threadIdx.y][threadIdx.x] = s1;
  __syncthreads();
  if (threadIdx.y == 0) {
#pragma unroll
    for (int y = 1; y < 8; ++y) s1 += red[y][threadIdx.x];
  }
  __syncthreads();
  red[threadIdx.y][threadIdx.x] = sT;
  __syncthreads();
  if (threadIdx.y == 0) {
#pragma unroll
    for (int y = 1; y < 8; ++y) sT += red[y][threadIdx.x];
    const int idx = (b * S_ + col) * 3;
    cstat[idx + 0] = m;
    cstat[idx + 1] = s1;
    cstat[idx + 2] = sT;
  }
}

// ---------------------------------------------------------------------------
// Kernel 5: fused finalize -> align_matrix (as float) and align_prob
// ---------------------------------------------------------------------------
__global__ __launch_bounds__(256) void finalize_kernel(const float* __restrict__ scores,
                                                       const float* __restrict__ add_src,
                                                       const float* __restrict__ add_tgt,
                                                       const float* __restrict__ len_src,
                                                       const float* __restrict__ len_tgt,
                                                       const float* __restrict__ rstat,
                                                       const float* __restrict__ cstat,
                                                       float* __restrict__ out_mat,
                                                       float* __restrict__ out_prob) {
  const size_t idx = (size_t)blockIdx.x * 256 + threadIdx.x;
  const int b = (int)(idx >> 20);
  const int s = (int)((idx >> 10) & (S_ - 1));
  const int t = (int)(idx & (S_ - 1));

  const float x = scores[idx];
  const float xs = x + add_tgt[b * S_ + t];  // masked for softmax over t
  const float xt = x + add_src[b * S_ + s];  // masked for softmax over s

  const float* rs = &rstat[(b * S_ + s) * 3];
  const float* cs = &cstat[(b * S_ + t) * 3];

  const float probs_src = __expf(xs - rs[0]) / rs[1];
  const float probs_tgt = __expf(xt - cs[0]) / cs[1];

  const float itT = rsqrtf(len_tgt[b]);
  const float itS = rsqrtf(len_src[b]);
  const float p_src = __expf((xs - rs[0]) * itT) / rs[2];
  const float p_tgt = __expf((xt - cs[0]) * itS) / cs[2];

  out_mat[idx] = (probs_src > 1.0e-3f && probs_tgt > 1.0e-3f) ? 1.0f : 0.0f;
  out_prob[idx] = 2.0f * p_src * p_tgt / (p_src + p_tgt + 1.0e-9f);
}

// ---------------------------------------------------------------------------
extern "C" void kernel_launch(void* const* d_in, const int* in_sizes, int n_in,
                              void* d_out, int out_size, void* d_ws, size_t ws_size,
                              hipStream_t stream) {
  const float* h_src = (const float*)d_in[0];
  const float* h_tgt = (const float*)d_in[1];
  const int* ids_src = (const int*)d_in[2];
  const int* ids_tgt = (const int*)d_in[3];

  float* ws = (float*)d_ws;
  float* scores  = ws;                                    // B*S*S (64 MB)
  float* add_src = scores + (size_t)B_ * S_ * S_;         // B*S
  float* add_tgt = add_src + B_ * S_;                     // B*S
  float* len_src = add_tgt + B_ * S_;                     // B
  float* len_tgt = len_src + B_;                          // B
  float* rstat   = len_tgt + B_;                          // 3*B*S
  float* cstat   = rstat + 3 * B_ * S_;                   // 3*B*S
  __bf16* src_bf = (__bf16*)(cstat + 3 * B_ * S_);        // B*S*D (32 MB)
  __bf16* tgt_bf = src_bf + (size_t)B_ * S_ * D_;         // B*S*D (32 MB)

  float* out_mat = (float*)d_out;
  float* out_prob = out_mat + (size_t)B_ * S_ * S_;

  const int cvt_blocks = (int)(((size_t)B_ * S_ * D_) / (256 * 8));
  cvt_f32_bf16_kernel<<<cvt_blocks, 256, 0, stream>>>(h_src, src_bf);
  cvt_f32_bf16_kernel<<<cvt_blocks, 256, 0, stream>>>(h_tgt, tgt_bf);

  mask_len_kernel<<<B_, 256, 0, stream>>>(ids_src, add_src, len_src);
  mask_len_kernel<<<B_, 256, 0, stream>>>(ids_tgt, add_tgt, len_tgt);

  gemm_scores_wmma<<<dim3(S_ / BN, S_ / BM, B_), 256, 0, stream>>>(src_bf, tgt_bf, scores);

  row_stats_kernel<<<B_ * S_, 256, 0, stream>>>(scores, add_tgt, len_tgt, rstat);
  col_stats_kernel<<<dim3(S_ / 32, B_), dim3(32, 8), 0, stream>>>(scores, add_src, len_src, cstat);

  finalize_kernel<<<(int)(((size_t)B_ * S_ * S_) / 256), 256, 0, stream>>>(
      scores, add_src, add_tgt, len_src, len_tgt, rstat, cstat, out_mat, out_prob);
}
